// CrossAttention_For_Fusion_Temporal_52853867544584
// MI455X (gfx1250) — matrix operand order repaired and checked by reading, not compiled, not run on hardware
//
#include <hip/hip_runtime.h>
#include <hip/hip_bf16.h>

typedef __bf16 bf16;
typedef __attribute__((ext_vector_type(16))) __bf16 v16bf;
typedef __attribute__((ext_vector_type(8)))  float v8f;
typedef unsigned int u32;
typedef __attribute__((ext_vector_type(4))) u32 u32x4;
typedef __attribute__((ext_vector_type(8))) u32 u32x8;

#define C_DIM 768
#define N_KEY 576
#define N_TOK 320
#define LN_EPS 1e-5f

__device__ __forceinline__ v8f vzero8() {
  v8f z;
#pragma unroll
  for (int r = 0; r < 8; ++r) z[r] = 0.0f;
  return z;
}

__device__ __forceinline__ v8f wmma_bf16(v16bf a, v16bf b, v8f c) {
  return __builtin_amdgcn_wmma_f32_16x16x32_bf16(false, a, false, b, (short)0, c,
                                                 false, false);
}

// A fragment (16x32 bf16) from row-major buffer, row stride ld.
// Lanes 0-15: M=lane, K=0..15; lanes 16-31: M=lane-16, K=16..31.
__device__ __forceinline__ v16bf load_a(const bf16* base, int ld, int lane) {
  const int m = lane & 15;
  const int hi = (lane >> 4) << 4;
  v16bf a;
#pragma unroll
  for (int j = 0; j < 8; ++j) {
    a[j]     = base[m * ld + hi + j];
    a[8 + j] = base[m * ld + hi + 8 + j];
  }
  return a;
}

// B fragment (32x16): source stored K-contiguous per output column
// (i.e. X^T where X is [N][K] row-major). Lane n -> one aligned 32B load.
__device__ __forceinline__ v16bf load_b_kcont(const bf16* base, int ld, int lane) {
  const int n = lane & 15;
  const int hi = (lane >> 4) << 4;
  const bf16* p = base + (size_t)n * ld + hi;
  return *(const v16bf*)p;
}

// ---- Tensor Data Mover: 1-D contiguous bf16 tile, global -> LDS -------------
// Builds the D# (groups 0/1) in SGPRs and issues TENSOR_LOAD_TO_LDS.
// nelem must fit tile_dim0 (16 bits). Issue from ONE wave only; pair with
// s_wait_tensorcnt + workgroup barrier before consumers read the LDS.
__device__ __forceinline__ void tdm_load_to_lds_bf16(const bf16* gsrc,
                                                     bf16* lds_dst, u32 nelem) {
  const unsigned long long ga = (unsigned long long)(uintptr_t)gsrc;
  const u32 la = (u32)(uintptr_t)lds_dst;  // LDS aperture lives in high 32 bits
  u32x4 g0;
  g0[0] = 1u;                                   // count=1, user descriptor
  g0[1] = la;                                   // lds_addr
  g0[2] = (u32)ga;                              // global_addr[31:0]
  g0[3] = (u32)(ga >> 32) | 0x80000000u;        // global_addr[56:32] | type=2
  u32x8 g1;
  g1[0] = 0x00010000u;                          // wg_mask=0, data_size=1 (2B)
  g1[1] = (nelem & 0xFFFFu) << 16;              // tensor_dim0[15:0]
  g1[2] = ((nelem >> 16) & 0xFFFFu) | (1u << 16);  // tensor_dim0[31:16], tensor_dim1=1
  g1[3] = nelem << 16;                          // tile_dim0
  g1[4] = 1u;                                   // tile_dim1=1, tile_dim2=0
  g1[5] = nelem;                                // tensor_dim0_stride[31:0]
  g1[6] = 0u;
  g1[7] = 0u;
  asm volatile("tensor_load_to_lds %0, %1" :: "s"(g0), "s"(g1) : "memory");
}

// ---------- transpose + convert: out_bf16[n][k] = (bf16)in_f32[k][n], 768x768 --
__global__ __launch_bounds__(256)
void transpose_cvt_768(const float* __restrict__ in, bf16* __restrict__ out) {
  __shared__ float tile[32][33];
  const int bx = blockIdx.x % 24, by = blockIdx.x / 24;
  const int tx = threadIdx.x & 31, ty = threadIdx.x >> 5;  // 32x8
#pragma unroll
  for (int r = 0; r < 32; r += 8)
    tile[ty + r][tx] = in[(size_t)(by * 32 + ty + r) * C_DIM + bx * 32 + tx];
  __syncthreads();
#pragma unroll
  for (int r = 0; r < 32; r += 8)
    out[(size_t)(bx * 32 + ty + r) * C_DIM + by * 32 + tx] = (bf16)tile[tx][ty + r];
}

// ---------------- Projection GEMM: out = cvt_bf16(in_f32 @ W), W given as WT ---
// Optionally also writes a transposed copy outT[c][row] (for PV B-fragments).
__global__ __launch_bounds__(256)
void proj_gemm_bf16(const float* __restrict__ in, const bf16* __restrict__ WT,
                    bf16* __restrict__ out, bf16* __restrict__ outT,
                    int rows_per_batch, int in_batch_stride, int in_row_offset) {
  __shared__ bf16 ldsA[16 * C_DIM];
  const int tiles_per_batch = rows_per_batch >> 4;
  const int b = blockIdx.x / tiles_per_batch;
  const int r0 = (blockIdx.x % tiles_per_batch) << 4;
  const int wv = threadIdx.x >> 5;
  const int lane = threadIdx.x & 31;

  const float* src = in + (size_t)b * in_batch_stride + in_row_offset +
                     (size_t)r0 * C_DIM;
  for (int idx = threadIdx.x; idx < 16 * C_DIM; idx += 256)
    ldsA[idx] = (bf16)src[idx];
  __syncthreads();

  v8f acc[6];
#pragma unroll
  for (int i = 0; i < 6; ++i) acc[i] = vzero8();

  for (int kk = 0; kk < C_DIM; kk += 32) {
    v16bf a = load_a(ldsA + kk, C_DIM, lane);
#pragma unroll
    for (int i = 0; i < 6; ++i) {
      const int n0 = (wv * 6 + i) << 4;
      v16bf bw = load_b_kcont(WT + (size_t)n0 * C_DIM + kk, C_DIM, lane);
      acc[i] = wmma_bf16(a, bw, acc[i]);
    }
  }

  const int col = lane & 15;
  const int rhalf = (lane >> 4) << 3;
  bf16* dst = out + ((size_t)b * rows_per_batch + r0) * C_DIM;
#pragma unroll
  for (int i = 0; i < 6; ++i) {
    const int n0 = (wv * 6 + i) << 4;
#pragma unroll
    for (int r = 0; r < 8; ++r)
      dst[(size_t)(rhalf + r) * C_DIM + n0 + col] = (bf16)acc[i][r];
  }
  if (outT) {
    bf16* dstT = outT + (size_t)b * C_DIM * rows_per_batch + r0;
#pragma unroll
    for (int i = 0; i < 6; ++i) {
      const int n0 = (wv * 6 + i) << 4;
#pragma unroll
      for (int r = 0; r < 8; ++r)
        dstT[(size_t)(n0 + col) * rows_per_batch + rhalf + r] = (bf16)acc[i][r];
    }
  }
}

// ------------- Fused attention: logits -> softmax -> counts -> PV -> Wout -> LN
__global__ __launch_bounds__(256)
void attn_fused(const bf16* __restrict__ qbf, const bf16* __restrict__ kbf,
                const bf16* __restrict__ kibf, const bf16* __restrict__ vT,
                const bf16* __restrict__ viT, const float* __restrict__ Tin,
                const bf16* __restrict__ WoutT, const float* __restrict__ gamma,
                const float* __restrict__ beta, int* __restrict__ g_pos,
                int* __restrict__ g_eq, float* __restrict__ outT) {
  __shared__ bf16 ldsU[2 * 16 * N_KEY];  // union: q tile / {P,Pi} / ctx (36.9 KB)
  __shared__ float red[2][8][16];
  __shared__ float rowstat[2][16];
  __shared__ int cntp[N_KEY];
  __shared__ int cnte[N_KEY];

  const int b = blockIdx.x / (N_TOK / 16);
  const int t0 = (blockIdx.x % (N_TOK / 16)) << 4;
  const int wv = threadIdx.x >> 5;
  const int lane = threadIdx.x & 31;
  const int col = lane & 15;
  const int rhalf = (lane >> 4) << 3;

  bf16* ldsQ = ldsU;
  bf16* ldsP0 = ldsU;
  bf16* ldsP1 = ldsU + 16 * N_KEY;
  bf16* ldsC = ldsU;

  // q tile: 16x768 contiguous bf16 -> LDS via Tensor Data Mover (wave 0 issues).
  {
    const bf16* qsrc = qbf + ((size_t)b * N_TOK + t0) * C_DIM;
    if (wv == 0) {
      tdm_load_to_lds_bf16(qsrc, ldsQ, 16 * C_DIM);
    }
    for (int idx = threadIdx.x; idx < N_KEY; idx += 256) { cntp[idx] = 0; cnte[idx] = 0; }
    if (wv == 0) __builtin_amdgcn_s_wait_tensorcnt(0);
  }
  __syncthreads();

  // ---- logits S = qk^T, Si = qki^T; 36 key tiles split over 8 waves ----
  v8f acc[2][5];
#pragma unroll
  for (int m = 0; m < 2; ++m)
#pragma unroll
    for (int i = 0; i < 5; ++i) acc[m][i] = vzero8();

  const bf16* kb = kbf + (size_t)b * N_KEY * C_DIM;
  const bf16* kib = kibf + (size_t)b * N_KEY * C_DIM;
  for (int kk = 0; kk < C_DIM; kk += 32) {
    v16bf a = load_a(ldsQ + kk, C_DIM, lane);
#pragma unroll
    for (int i = 0; i < 5; ++i) {
      const int nt = wv + (i << 3);
      if (nt < 36) {
        v16bf b0 = load_b_kcont(kb + ((size_t)nt * 16) * C_DIM + kk, C_DIM, lane);
        v16bf b1 = load_b_kcont(kib + ((size_t)nt * 16) * C_DIM + kk, C_DIM, lane);
        acc[0][i] = wmma_bf16(a, b0, acc[0][i]);
        acc[1][i] = wmma_bf16(a, b1, acc[1][i]);
      }
    }
  }

  const float scale = 0.03608439182435161f;  // 768^-0.5
#pragma unroll
  for (int m = 0; m < 2; ++m)
#pragma unroll
    for (int i = 0; i < 5; ++i) acc[m][i] = acc[m][i] * scale;

  // ---- softmax across waves (row max, then row sum) ----
#pragma unroll
  for (int m = 0; m < 2; ++m) {
    v8f mx;
#pragma unroll
    for (int r = 0; r < 8; ++r) mx[r] = -3.0e38f;
#pragma unroll
    for (int i = 0; i < 5; ++i)
      if (wv + (i << 3) < 36)
#pragma unroll
        for (int r = 0; r < 8; ++r) mx[r] = fmaxf(mx[r], acc[m][i][r]);
#pragma unroll
    for (int r = 0; r < 8; ++r) {
      float v = mx[r];
      for (int off = 1; off < 16; off <<= 1) v = fmaxf(v, __shfl_xor(v, off, 32));
      mx[r] = v;
    }
    if (col == 0)
#pragma unroll
      for (int r = 0; r < 8; ++r) red[m][wv][rhalf + r] = mx[r];
  }
  __syncthreads();
  if (threadIdx.x < 32) {
    const int m = threadIdx.x >> 4, row = threadIdx.x & 15;
    float g = red[m][0][row];
#pragma unroll
    for (int w = 1; w < 8; ++w) g = fmaxf(g, red[m][w][row]);
    rowstat[m][row] = g;
  }
  __syncthreads();

#pragma unroll
  for (int m = 0; m < 2; ++m) {
    v8f sm = vzero8();
#pragma unroll
    for (int i = 0; i < 5; ++i)
      if (wv + (i << 3) < 36)
#pragma unroll
        for (int r = 0; r < 8; ++r) {
          float p = __expf(acc[m][i][r] - rowstat[m][rhalf + r]);
          acc[m][i][r] = p;
          sm[r] += p;
        }
#pragma unroll
    for (int r = 0; r < 8; ++r) {
      float v = sm[r];
      for (int off = 1; off < 16; off <<= 1) v += __shfl_xor(v, off, 32);
      sm[r] = v;
    }
    if (col == 0)
#pragma unroll
      for (int r = 0; r < 8; ++r) red[m][wv][rhalf + r] = sm[r];
  }
  __syncthreads();
  if (threadIdx.x < 32) {
    const int m = threadIdx.x >> 4, row = threadIdx.x & 15;
    float s = 0.f;
#pragma unroll
    for (int w = 0; w < 8; ++w) s += red[m][w][row];
    rowstat[m][row] = 1.0f / s;
  }
  __syncthreads();

  // normalize, stash P/Pi into LDS (q tile is dead), tally sign counts
#pragma unroll
  for (int i = 0; i < 5; ++i) {
    const int nt = wv + (i << 3);
    if (nt < 36) {
#pragma unroll
      for (int r = 0; r < 8; ++r) {
        const int row = rhalf + r;
        const int n = (nt << 4) + col;
        float p = acc[0][i][r] * rowstat[0][row];
        float pi = acc[1][i][r] * rowstat[1][row];
        ldsP0[row * N_KEY + n] = (bf16)p;
        ldsP1[row * N_KEY + n] = (bf16)pi;
        if (p > pi) atomicAdd(&cntp[n], 1);
        else if (p == pi) atomicAdd(&cnte[n], 1);
      }
    }
  }
  __syncthreads();

  for (int n = threadIdx.x; n < N_KEY; n += 256) {
    if (cntp[n]) atomicAdd(&g_pos[b * N_KEY + n], cntp[n]);
    if (cnte[n]) atomicAdd(&g_eq[b * N_KEY + n], cnte[n]);
  }

  // ---- ctx = P V + Pi Vi ; B-fragments from vT (K-contiguous) ----
  v8f ctx[6];
#pragma unroll
  for (int i = 0; i < 6; ++i) ctx[i] = vzero8();
  const bf16* vTb = vT + (size_t)b * C_DIM * N_KEY;
  const bf16* viTb = viT + (size_t)b * C_DIM * N_KEY;
  for (int kk = 0; kk < N_KEY; kk += 32) {
    v16bf aP = load_a(ldsP0 + kk, N_KEY, lane);
    v16bf aPi = load_a(ldsP1 + kk, N_KEY, lane);
#pragma unroll
    for (int i = 0; i < 6; ++i) {
      const int n0 = (wv * 6 + i) << 4;
      v16bf bv = load_b_kcont(vTb + (size_t)n0 * N_KEY + kk, N_KEY, lane);
      v16bf bvi = load_b_kcont(viTb + (size_t)n0 * N_KEY + kk, N_KEY, lane);
      ctx[i] = wmma_bf16(aP, bv, ctx[i]);
      ctx[i] = wmma_bf16(aPi, bvi, ctx[i]);
    }
  }
  __syncthreads();  // everyone is done reading P/Pi
#pragma unroll
  for (int i = 0; i < 6; ++i) {
    const int n0 = (wv * 6 + i) << 4;
#pragma unroll
    for (int r = 0; r < 8; ++r)
      ldsC[(rhalf + r) * C_DIM + n0 + col] = (bf16)ctx[i][r];
  }
  __syncthreads();

  // ---- y = T + ctx @ Wout (B from WoutT, K-contiguous) ----
  v8f y[6];
#pragma unroll
  for (int i = 0; i < 6; ++i) y[i] = vzero8();
  for (int kk = 0; kk < C_DIM; kk += 32) {
    v16bf a = load_a(ldsC + kk, C_DIM, lane);
#pragma unroll
    for (int i = 0; i < 6; ++i) {
      const int n0 = (wv * 6 + i) << 4;
      v16bf bw = load_b_kcont(WoutT + (size_t)n0 * C_DIM + kk, C_DIM, lane);
      y[i] = wmma_bf16(a, bw, y[i]);
    }
  }
  const float* Trow = Tin + ((size_t)b * N_TOK + t0) * C_DIM;
#pragma unroll
  for (int i = 0; i < 6; ++i) {
    const int n0 = (wv * 6 + i) << 4;
#pragma unroll
    for (int r = 0; r < 8; ++r)
      y[i][r] += Trow[(size_t)(rhalf + r) * C_DIM + n0 + col];
  }

  // ---- LayerNorm over the 768-wide rows owned by this block ----
  v8f ps = vzero8(), ps2 = vzero8();
#pragma unroll
  for (int i = 0; i < 6; ++i) { ps += y[i]; ps2 += y[i] * y[i]; }
#pragma unroll
  for (int r = 0; r < 8; ++r) {
    float a1 = ps[r], a2 = ps2[r];
    for (int off = 1; off < 16; off <<= 1) {
      a1 += __shfl_xor(a1, off, 32);
      a2 += __shfl_xor(a2, off, 32);
    }
    ps[r] = a1; ps2[r] = a2;
  }
  if (col == 0)
#pragma unroll
    for (int r = 0; r < 8; ++r) {
      red[0][wv][rhalf + r] = ps[r];
      red[1][wv][rhalf + r] = ps2[r];
    }
  __syncthreads();
  if (threadIdx.x < 32) {
    const int m = threadIdx.x >> 4, row = threadIdx.x & 15;
    float s = 0.f;
#pragma unroll
    for (int w = 0; w < 8; ++w) s += red[m][w][row];
    rowstat[m][row] = s * (1.0f / C_DIM);
  }
  __syncthreads();

  float* orow = outT + ((size_t)b * N_TOK + t0) * C_DIM;
#pragma unroll
  for (int i = 0; i < 6; ++i) {
    const int n0 = (wv * 6 + i) << 4;
#pragma unroll
    for (int r = 0; r < 8; ++r) {
      const int row = rhalf + r;
      const float mu = rowstat[0][row];
      const float var = rowstat[1][row] - mu * mu;
      const float rstd = rsqrtf(var + LN_EPS);
      const int c = n0 + col;
      orow[(size_t)row * C_DIM + c] = (y[i][r] - mu) * rstd * gamma[c] + beta[c];
    }
  }
}

// ------------- x_s = v*(rgb+eq) + vi*(1-rgb) -------------
__global__ __launch_bounds__(256)
void xs_kernel(const bf16* __restrict__ vbf, const bf16* __restrict__ vibf,
               const int* __restrict__ g_pos, const int* __restrict__ g_eq,
               float* __restrict__ out) {
  const int row = blockIdx.x;  // b*576 + n
  const float inv_rows = 1.0f / 320.0f;
  const float rgb = (float)g_pos[row] * inv_rows;
  const float eq = (float)g_eq[row] * inv_rows;
  const float r1 = rgb + eq;
  const float r2 = 1.0f - rgb;
  const bf16* v = vbf + (size_t)row * C_DIM;
  const bf16* vi = vibf + (size_t)row * C_DIM;
  float* o = out + (size_t)row * C_DIM;
  for (int c = threadIdx.x; c < C_DIM; c += 256)
    o[c] = (float)v[c] * r1 + (float)vi[c] * r2;
}

extern "C" void kernel_launch(void* const* d_in, const int* in_sizes, int n_in,
                              void* d_out, int out_size, void* d_ws, size_t ws_size,
                              hipStream_t stream) {
  (void)in_sizes; (void)n_in; (void)out_size; (void)ws_size;
  const float* T    = (const float*)d_in[0];
  const float* x    = (const float*)d_in[1];
  const float* xi   = (const float*)d_in[2];
  const float* Wq   = (const float*)d_in[3];
  const float* Wk   = (const float*)d_in[4];
  const float* Wv   = (const float*)d_in[5];
  const float* Wout = (const float*)d_in[6];
  const float* gamma = (const float*)d_in[7];
  const float* beta  = (const float*)d_in[8];

  char* ws = (char*)d_ws;
  const size_t SZ_Q = (size_t)32 * N_TOK * C_DIM * sizeof(bf16);
  const size_t SZ_KV = (size_t)32 * N_KEY * C_DIM * sizeof(bf16);
  const size_t SZ_W = (size_t)C_DIM * C_DIM * sizeof(bf16);
  size_t off = 0;
  bf16* qb   = (bf16*)(ws + off); off += SZ_Q;
  bf16* kb   = (bf16*)(ws + off); off += SZ_KV;
  bf16* kib  = (bf16*)(ws + off); off += SZ_KV;
  bf16* vb   = (bf16*)(ws + off); off += SZ_KV;
  bf16* vib  = (bf16*)(ws + off); off += SZ_KV;
  bf16* vTb  = (bf16*)(ws + off); off += SZ_KV;
  bf16* viTb = (bf16*)(ws + off); off += SZ_KV;
  bf16* WqT  = (bf16*)(ws + off); off += SZ_W;
  bf16* WkT  = (bf16*)(ws + off); off += SZ_W;
  bf16* WvT  = (bf16*)(ws + off); off += SZ_W;
  bf16* WoT  = (bf16*)(ws + off); off += SZ_W;
  int* g_pos = (int*)(ws + off);
  int* g_eq  = g_pos + 32 * N_KEY;

  hipMemsetAsync(g_pos, 0, 2 * 32 * N_KEY * sizeof(int), stream);

  // Weights -> bf16, transposed to K-contiguous layout (done every call; tiny)
  transpose_cvt_768<<<576, 256, 0, stream>>>(Wq, WqT);
  transpose_cvt_768<<<576, 256, 0, stream>>>(Wk, WkT);
  transpose_cvt_768<<<576, 256, 0, stream>>>(Wv, WvT);
  transpose_cvt_768<<<576, 256, 0, stream>>>(Wout, WoT);

  // Projections (bf16 outputs; v/vi also emit transposed copies for PV)
  proj_gemm_bf16<<<32 * (N_TOK / 16), 256, 0, stream>>>(
      T, WqT, qb, nullptr, N_TOK, N_TOK * C_DIM, 0);
  proj_gemm_bf16<<<32 * (N_KEY / 16), 256, 0, stream>>>(
      x, WkT, kb, nullptr, N_KEY, 704 * C_DIM, 128 * C_DIM);
  proj_gemm_bf16<<<32 * (N_KEY / 16), 256, 0, stream>>>(
      xi, WkT, kib, nullptr, N_KEY, 704 * C_DIM, 128 * C_DIM);
  proj_gemm_bf16<<<32 * (N_KEY / 16), 256, 0, stream>>>(
      x, WvT, vb, vTb, N_KEY, 704 * C_DIM, 128 * C_DIM);
  proj_gemm_bf16<<<32 * (N_KEY / 16), 256, 0, stream>>>(
      xi, WvT, vib, viTb, N_KEY, 704 * C_DIM, 128 * C_DIM);

  float* outT = (float*)d_out + (size_t)32 * N_KEY * C_DIM;
  attn_fused<<<32 * (N_TOK / 16), 256, 0, stream>>>(qb, kb, kib, vTb, viTb, T, WoT,
                                                    gamma, beta, g_pos, g_eq, outT);
  xs_kernel<<<32 * N_KEY, 256, 0, stream>>>(vb, vib, g_pos, g_eq, (float*)d_out);
}
